// TemporalRSSM_43224550866995
// MI455X (gfx1250) — compile-verified, matrix operand order checked by reading, not run on hardware
//
#include <hip/hip_runtime.h>
#include <hip/hip_bf16.h>
#include <stdint.h>

// ---------------------------------------------------------------------------
// TemporalRSSM for MI455X (gfx1250)
//   bf16 WMMA GEMM pipeline with async-DMA (global->LDS) double buffering,
//   lane-parallel selective scan, per-token head.
// ---------------------------------------------------------------------------

typedef __attribute__((ext_vector_type(16))) __bf16   v16bf;
typedef __attribute__((ext_vector_type(8)))  float    v8f;
typedef __attribute__((ext_vector_type(4)))  unsigned u32x4;

#define B_      8
#define T_      4096
#define D_      256
#define INNER_  512
#define S_      16
#define NT_     (B_ * T_)      // 32768 tokens
#define NCODES_ 240

// ---------------- activations ----------------
__device__ __forceinline__ float sigmoidf_(float x) {
    return 1.f / (1.f + __expf(-x));
}
__device__ __forceinline__ float apply_act(float x, int act) {
    if (act == 1) return x * sigmoidf_(x);                          // silu
    if (act == 2) return (x > 20.f) ? x : log1pf(__expf(x));        // softplus
    if (act == 3) {                                                 // gelu (tanh approx)
        float x3 = x * x * x;
        return 0.5f * x * (1.f + tanhf(0.7978845608028654f * (x + 0.044715f * x3)));
    }
    return x;
}

// ---------------- packing pre-passes ----------------
__global__ __launch_bounds__(256)
void f32_to_bf16_kernel(const float* __restrict__ src, __bf16* __restrict__ dst, long n) {
    long i = (long)blockIdx.x * 256 + threadIdx.x;
    long stride = (long)gridDim.x * 256;
    for (; i < n; i += stride) dst[i] = (__bf16)src[i];
}

// W[K,N] fp32 -> Wt[N,K] bf16 (so GEMM B-tiles are K-contiguous per column)
__global__ __launch_bounds__(256)
void transpose_to_bf16_kernel(const float* __restrict__ W, __bf16* __restrict__ Wt,
                              int K, int N) {
    long i = (long)blockIdx.x * 256 + threadIdx.x;
    if (i < (long)K * N) {
        int k = (int)(i / N), n = (int)(i % N);
        Wt[(long)n * K + k] = (__bf16)W[i];
    }
}

// ---------------- WMMA bf16 GEMM ----------------
// C = act(A[M,K](bf16) @ W[K,N] + bias), W given pre-transposed Wt[N,K] bf16.
// Block tile 64x128, 8 waves each owning 32x32 (4 wmma / K-step of 32).
// Tiles staged with GLOBAL_LOAD_ASYNC_TO_LDS_B128 + double buffering.
#define BM 64
#define BN 128
#define BK 32
#define LDK 40                 // padded row: 80B = 20 banks -> conflict-free, 16B aligned

__device__ __forceinline__ void async_copy16(uint32_t lds_dst, const __bf16* gsrc) {
    uint64_t ga = (uint64_t)(uintptr_t)gsrc;
    asm volatile("global_load_async_to_lds_b128 %0, %1, off"
                 :: "v"(lds_dst), "v"(ga) : "memory");
}

__global__ __launch_bounds__(256)
void gemm_bf16_wmma(const __bf16* __restrict__ A, long lda,
                    const __bf16* __restrict__ Wt,      // [N,K]
                    const float* __restrict__ bias,
                    float*  __restrict__ Cf,            // optional fp32 out
                    __bf16* __restrict__ Cb,            // optional bf16 out
                    long ldc, int M, int N, int K, int act)
{
    __shared__ __align__(16) __bf16 As[2][BM][LDK];
    __shared__ __align__(16) __bf16 Bs[2][BN][LDK];

    const int tid  = threadIdx.x;
    const int wave = tid >> 5;
    const int lane = tid & 31;
    const int half = lane >> 4;          // K halves {0-7,16-23} / {8-15,24-31}
    const int lm   = lane & 15;
    const int wm   = wave >> 2;          // 0..1
    const int wn   = wave & 3;           // 0..3
    const long row0 = (long)blockIdx.x * BM;
    const int  col0 = blockIdx.y * BN;

    // staging geometry: each thread moves one 16B chunk of A and two of Wt
    const int sm = tid >> 2;             // 0..63 (A row / Wt row base)
    const int sk = (tid & 3) * 8;        // K chunk offset

    auto stage = [&](int k0, int buf) {
        async_copy16((uint32_t)(uintptr_t)&As[buf][sm][sk],
                     A + (row0 + sm) * lda + (k0 + sk));
        #pragma unroll
        for (int j = 0; j < 2; ++j) {
            int n = sm + 64 * j;
            if (col0 + n < N)
                async_copy16((uint32_t)(uintptr_t)&Bs[buf][n][sk],
                             Wt + (long)(col0 + n) * K + (k0 + sk));
        }
    };

    v8f acc00 = {}, acc01 = {}, acc10 = {}, acc11 = {};

    const int nk = K / BK;
    stage(0, 0);
    for (int it = 0; it < nk; ++it) {
        const int buf = it & 1;
        asm volatile("s_wait_asynccnt 0x0" ::: "memory");
        __syncthreads();                      // tile[buf] visible to all waves;
                                              // tile[buf^1] no longer being read
        if (it + 1 < nk) stage((it + 1) * BK, buf ^ 1);

        // fragment loads: two ds_load_b128 per v16bf (ISA 7.12.2 layout)
        union Frag { u32x4 q[2]; v16bf v; };
        Frag a0, a1, b0, b1;
        {
            const __bf16* pa0 = &As[buf][wm * 32 + lm][half * 8];
            const __bf16* pa1 = &As[buf][wm * 32 + 16 + lm][half * 8];
            const __bf16* pb0 = &Bs[buf][wn * 32 + lm][half * 8];
            const __bf16* pb1 = &Bs[buf][wn * 32 + 16 + lm][half * 8];
            a0.q[0] = *(const u32x4*)__builtin_assume_aligned(pa0, 16);
            a0.q[1] = *(const u32x4*)__builtin_assume_aligned(pa0 + 16, 16);
            a1.q[0] = *(const u32x4*)__builtin_assume_aligned(pa1, 16);
            a1.q[1] = *(const u32x4*)__builtin_assume_aligned(pa1 + 16, 16);
            b0.q[0] = *(const u32x4*)__builtin_assume_aligned(pb0, 16);
            b0.q[1] = *(const u32x4*)__builtin_assume_aligned(pb0 + 16, 16);
            b1.q[0] = *(const u32x4*)__builtin_assume_aligned(pb1, 16);
            b1.q[1] = *(const u32x4*)__builtin_assume_aligned(pb1 + 16, 16);
        }
        acc00 = __builtin_amdgcn_wmma_f32_16x16x32_bf16(false, a0.v, false, b0.v, (short)0, acc00, false, false);
        acc01 = __builtin_amdgcn_wmma_f32_16x16x32_bf16(false, a0.v, false, b1.v, (short)0, acc01, false, false);
        acc10 = __builtin_amdgcn_wmma_f32_16x16x32_bf16(false, a1.v, false, b0.v, (short)0, acc10, false, false);
        acc11 = __builtin_amdgcn_wmma_f32_16x16x32_bf16(false, a1.v, false, b1.v, (short)0, acc11, false, false);
    }

    // epilogue: C/D layout — VGPR r -> rows half*8+r, col = lane%16 of 16-col group
    #pragma unroll
    for (int mi = 0; mi < 2; ++mi) {
        #pragma unroll
        for (int ni = 0; ni < 2; ++ni) {
            v8f accv = (mi == 0) ? (ni == 0 ? acc00 : acc01)
                                 : (ni == 0 ? acc10 : acc11);
            int col = col0 + wn * 32 + ni * 16 + lm;
            if (col < N) {
                float bv = bias ? bias[col] : 0.f;
                #pragma unroll
                for (int r = 0; r < 8; ++r) {
                    long row = row0 + wm * 32 + mi * 16 + half * 8 + r;
                    float v = apply_act(accv[r] + bv, act);
                    if (Cf) Cf[row * ldc + col] = v;
                    if (Cb) Cb[row * ldc + col] = (__bf16)v;
                }
            }
        }
    }
}

// ---------------- selective scan ----------------
// one lane per (channel, state): 4096 channels x 16 states = 65536 lanes;
// width-16 xor-shuffle reduction for y_t; silu gate fused. bf16 inputs.
__global__ __launch_bounds__(256)
void scan_kernel(const __bf16* __restrict__ xconv,   // [NT,512]
                 const __bf16* __restrict__ dtb,     // [NT,512] softplus'd
                 const __bf16* __restrict__ bcb,     // [NT,32]  (B|C)
                 const __bf16* __restrict__ xp,      // [NT,1024], gate at 512+
                 const float*  __restrict__ A_log,   // [512,16]
                 __bf16* __restrict__ yscan)         // [NT,512]
{
    int gtid = blockIdx.x * 256 + threadIdx.x;
    int ci = gtid >> 4;           // channel 0..4095
    int s  = gtid & 15;
    int b  = ci >> 9;
    int i  = ci & 511;

    float Aval = -__expf(A_log[i * 16 + s]);
    float h = 0.f;
    long base = (long)b * T_;

    for (int t = 0; t < T_; ++t) {
        long tok = base + t;
        float Bv  = (float)bcb[tok * 32 + s];
        float Cv  = (float)bcb[tok * 32 + 16 + s];
        float xv  = (float)xconv[tok * 512 + i];
        float dtv = (float)dtb[tok * 512 + i];
        float ab  = __expf(dtv * Aval);
        h = ab * h + (dtv * Bv) * xv;
        float yp = h * Cv;
        yp += __shfl_xor(yp, 1, 16);
        yp += __shfl_xor(yp, 2, 16);
        yp += __shfl_xor(yp, 4, 16);
        yp += __shfl_xor(yp, 8, 16);
        if (s == 0) {
            float g = (float)xp[tok * 1024 + 512 + i];
            yscan[tok * 512 + i] = (__bf16)(yp * (g * sigmoidf_(g)));
        }
    }
}

// ---------------- per-token head ----------------
__global__ __launch_bounds__(256)
void head_kernel(const float*  __restrict__ y,       // [NT,256] fp32 (d_out)
                 const __bf16* __restrict__ f1,      // [NT,64] gelu(cat1)
                 const __bf16* __restrict__ e1,      // [NT,64] gelu(e8a)
                 const float* __restrict__ cat2_w, const float* __restrict__ cat2_b,
                 const float* __restrict__ risk_w, const float* __restrict__ risk_b,
                 const float* __restrict__ e8b_w,  const float* __restrict__ e8b_b,
                 const float* __restrict__ codebook,
                 float* __restrict__ e8_out, float* __restrict__ idx_out,
                 float* __restrict__ bif_out)
{
    int wave = threadIdx.x >> 5;
    int lane = threadIdx.x & 31;
    long tok = (long)blockIdx.x * 8 + wave;
    int t = (int)(tok % T_);

    // risk head
    float f2 = cat2_b[lane];
    const __bf16* f1t = f1 + tok * 64;
    #pragma unroll 8
    for (int k = 0; k < 64; ++k) f2 += (float)f1t[k] * cat2_w[k * 32 + lane];
    float rsum = f2 * risk_w[lane];
    #pragma unroll
    for (int off = 16; off; off >>= 1) rsum += __shfl_xor(rsum, off, 32);
    float risk = sigmoidf_(rsum + risk_b[0]);

    // velocity / acceleration
    const float* yt = y + tok * 256;
    float vsq = 0.f, vsqp = 0.f;
    for (int c = lane; c < 256; c += 32) {
        float cur   = yt[c];
        float prev  = (t >= 1) ? yt[c - 256] : cur;
        float prev2 = (t >= 2) ? yt[c - 512] : prev;
        float d1 = cur - prev;   vsq  += d1 * d1;
        float d2 = prev - prev2; vsqp += d2 * d2;
    }
    #pragma unroll
    for (int off = 16; off; off >>= 1) {
        vsq  += __shfl_xor(vsq,  off, 32);
        vsqp += __shfl_xor(vsqp, off, 32);
    }
    float vn = sqrtf(vsq), vnp = sqrtf(vsqp);
    float acc = fabsf(vn - vnp);
    float combined = 0.5f * risk + 0.3f * sigmoidf_(vn) + 0.2f * sigmoidf_(acc);

    // E8 quantizer
    float e8c[8];
    #pragma unroll
    for (int j = 0; j < 8; ++j) e8c[j] = e8b_b[j];
    const __bf16* e1t = e1 + tok * 64;
    for (int k = 0; k < 64; ++k) {
        float ev = (float)e1t[k];
        #pragma unroll
        for (int j = 0; j < 8; ++j) e8c[j] += ev * e8b_w[k * 8 + j];
    }
    float dmin = 3.4e38f;
    int   imin = 0;
    for (int c = lane; c < NCODES_; c += 32) {
        const float* cb = codebook + c * 8;
        float d = 0.f;
        #pragma unroll
        for (int j = 0; j < 8; ++j) { float df = e8c[j] - cb[j]; d += df * df; }
        if (d < dmin) { dmin = d; imin = c; }
    }
    #pragma unroll
    for (int off = 16; off; off >>= 1) {
        float od = __shfl_xor(dmin, off, 32);
        int   oi = __shfl_xor(imin, off, 32);
        if (od < dmin || (od == dmin && oi < imin)) { dmin = od; imin = oi; }
    }

    if (lane < 8) e8_out[tok * 8 + lane] = codebook[imin * 8 + lane];
    if (lane == 0) {
        idx_out[tok] = (float)imin;
        bif_out[tok] = (combined > 0.7f) ? 1.f : 0.f;
    }
}

// ---------------------------------------------------------------------------
extern "C" void kernel_launch(void* const* d_in, const int* in_sizes, int n_in,
                              void* d_out, int out_size, void* d_ws, size_t ws_size,
                              hipStream_t stream) {
    const float* x        = (const float*)d_in[0];
    const float* in_w     = (const float*)d_in[1];
    const float* in_b     = (const float*)d_in[2];
    const float* conv_w   = (const float*)d_in[3];
    const float* conv_b   = (const float*)d_in[4];
    const float* A_log    = (const float*)d_in[5];
    const float* bc_w     = (const float*)d_in[6];
    const float* bc_b     = (const float*)d_in[7];
    const float* dt_w     = (const float*)d_in[8];
    const float* dt_b     = (const float*)d_in[9];
    const float* out_w    = (const float*)d_in[10];
    const float* out_b    = (const float*)d_in[11];
    const float* cat1_w   = (const float*)d_in[12];
    const float* cat1_b   = (const float*)d_in[13];
    const float* cat2_w   = (const float*)d_in[14];
    const float* cat2_b   = (const float*)d_in[15];
    const float* risk_w   = (const float*)d_in[16];
    const float* risk_b   = (const float*)d_in[17];
    const float* e8a_w    = (const float*)d_in[18];
    const float* e8a_b    = (const float*)d_in[19];
    const float* e8b_w    = (const float*)d_in[20];
    const float* e8b_b    = (const float*)d_in[21];
    const float* codebook = (const float*)d_in[22];

    // ---- workspace carve-up (bf16 buffers; ~205 MB total) ----
    char* w = (char*)d_ws;
    auto alloc = [&](size_t bytes) { void* p = (void*)w; w += (bytes + 255) & ~(size_t)255; return p; };
    __bf16* xbf     = (__bf16*)alloc((size_t)NT_ * 256 * 2);
    __bf16* xp_bf   = (__bf16*)alloc((size_t)NT_ * 1024 * 2);
    __bf16* xconvbf = (__bf16*)alloc((size_t)NT_ * 512 * 2);
    __bf16* dtbf    = (__bf16*)alloc((size_t)NT_ * 512 * 2);
    __bf16* bcbf    = (__bf16*)alloc((size_t)NT_ * 32 * 2);
    __bf16* yscanbf = (__bf16*)alloc((size_t)NT_ * 512 * 2);
    __bf16* ybf     = (__bf16*)alloc((size_t)NT_ * 256 * 2);
    __bf16* f1bf    = (__bf16*)alloc((size_t)NT_ * 64 * 2);
    __bf16* e1bf    = (__bf16*)alloc((size_t)NT_ * 64 * 2);
    __bf16* in_wt   = (__bf16*)alloc((size_t)256 * 1024 * 2);
    __bf16* conv_wt = (__bf16*)alloc((size_t)512 * 512 * 2);
    __bf16* bc_wt   = (__bf16*)alloc((size_t)512 * 32 * 2);
    __bf16* dt_wt   = (__bf16*)alloc((size_t)512 * 512 * 2);
    __bf16* out_wt  = (__bf16*)alloc((size_t)512 * 256 * 2);
    __bf16* cat1_wt = (__bf16*)alloc((size_t)256 * 64 * 2);
    __bf16* e8a_wt  = (__bf16*)alloc((size_t)256 * 64 * 2);

    // d_out layout: y | e8_codes | idx | is_bif
    float* yout = (float*)d_out;
    float* e8o  = yout + (size_t)NT_ * 256;
    float* idxo = e8o  + (size_t)NT_ * 8;
    float* bifo = idxo + (size_t)NT_;

    dim3 blk(256);
    auto grd  = [](int M, int N) { return dim3(M / BM, (N + BN - 1) / BN); };
    auto tgrd = [](int K, int N) { return dim3(((long)K * N + 255) / 256); };

    // ---- pack inputs to bf16 (x) and transposed-bf16 weights ----
    f32_to_bf16_kernel<<<1024, blk, 0, stream>>>(x, xbf, (long)NT_ * 256);
    transpose_to_bf16_kernel<<<tgrd(256, 1024), blk, 0, stream>>>(in_w,   in_wt,   256, 1024);
    transpose_to_bf16_kernel<<<tgrd(512, 512),  blk, 0, stream>>>(conv_w, conv_wt, 512, 512);
    transpose_to_bf16_kernel<<<tgrd(512, 32),   blk, 0, stream>>>(bc_w,   bc_wt,   512, 32);
    transpose_to_bf16_kernel<<<tgrd(512, 512),  blk, 0, stream>>>(dt_w,   dt_wt,   512, 512);
    transpose_to_bf16_kernel<<<tgrd(512, 256),  blk, 0, stream>>>(out_w,  out_wt,  512, 256);
    transpose_to_bf16_kernel<<<tgrd(256, 64),   blk, 0, stream>>>(cat1_w, cat1_wt, 256, 64);
    transpose_to_bf16_kernel<<<tgrd(256, 64),   blk, 0, stream>>>(e8a_w,  e8a_wt,  256, 64);

    // 1) xp = x @ in_proj + b                        [NT,1024] bf16
    gemm_bf16_wmma<<<grd(NT_, 1024), blk, 0, stream>>>(xbf, 256, in_wt, in_b, nullptr, xp_bf, 1024, NT_, 1024, 256, 0);
    // 2) x_conv = silu(x_main @ conv_proj + b)       [NT,512] bf16 (x_main = xp cols 0..511)
    gemm_bf16_wmma<<<grd(NT_, 512), blk, 0, stream>>>(xp_bf, 1024, conv_wt, conv_b, nullptr, xconvbf, 512, NT_, 512, 512, 1);
    // 3) bc = x_conv @ bc_proj + b                   [NT,32] bf16
    gemm_bf16_wmma<<<grd(NT_, 32), blk, 0, stream>>>(xconvbf, 512, bc_wt, bc_b, nullptr, bcbf, 32, NT_, 32, 512, 0);
    // 4) dt = softplus(x_conv @ dt_proj + b)         [NT,512] bf16
    gemm_bf16_wmma<<<grd(NT_, 512), blk, 0, stream>>>(xconvbf, 512, dt_wt, dt_b, nullptr, dtbf, 512, NT_, 512, 512, 2);
    // 5) selective scan + silu gate                  [NT,512] bf16
    scan_kernel<<<256, blk, 0, stream>>>(xconvbf, dtbf, bcbf, xp_bf, A_log, yscanbf);
    // 6) y = yscan @ out_proj + b                    fp32 -> d_out, bf16 -> heads
    gemm_bf16_wmma<<<grd(NT_, 256), blk, 0, stream>>>(yscanbf, 512, out_wt, out_b, yout, ybf, 256, NT_, 256, 512, 0);
    // 7) f1 = gelu(y @ cat1 + b)                     [NT,64] bf16
    gemm_bf16_wmma<<<grd(NT_, 64), blk, 0, stream>>>(ybf, 256, cat1_wt, cat1_b, nullptr, f1bf, 64, NT_, 64, 256, 3);
    // 8) e1 = gelu(y @ e8a + b)                      [NT,64] bf16
    gemm_bf16_wmma<<<grd(NT_, 64), blk, 0, stream>>>(ybf, 256, e8a_wt, e8a_b, nullptr, e1bf, 64, NT_, 64, 256, 3);
    // 9) risk / bifurcation / E8 quantization
    head_kernel<<<NT_ / 8, blk, 0, stream>>>(yout, f1bf, e1bf, cat2_w, cat2_b, risk_w, risk_b,
                                             e8b_w, e8b_b, codebook, e8o, idxo, bifo);
}
